// WITRAN_2DPSGMU_Encoder_39565238731430
// MI455X (gfx1250) — matrix operand order repaired and checked by reading, not compiled, not run on hardware
//
#include <hip/hip_runtime.h>
#include <hip/hip_bf16.h>

// ---------------- problem constants (match reference) ----------------
#define HID     128
#define NL      3
#define RROWS   32
#define OCOLS   48
#define BATCH   64
#define FIN     32
#define LSTEPS  79          // RROWS + OCOLS - 1
#define RB      2048        // RROWS * BATCH
#define NWIDTH  768         // 6*HID
#define K0      288         // FIN + 2*HID   (layer 0)
#define K1      512         // 4*HID         (layers 1,2)

#define NBLOCKS   64        // RB / 32  (M-tile = 32 rows per block)
#define NTHREADS  512       // 16 waves of 32
#define AS        40        // LDS row stride (bf16) for A tile: 80B rows, 16B-aligned frags
#define BS        40        // LDS row stride (bf16) for B tile: conflict-free b128 reads

#define OUT_SLICE_ELEMS  (RB * LSTEPS * 2 * HID)                 // 41,418,752
#define ROW_SAVE_OFF     OUT_SLICE_ELEMS
#define COL_SAVE_OFF     (ROW_SAVE_OFF + BATCH * NL * RROWS * HID) // +786,432

typedef __bf16 v16bf __attribute__((ext_vector_type(16)));
typedef float  v8f   __attribute__((ext_vector_type(8)));

union FragU { v16bf v; uint4 q[2]; unsigned u[8]; };

__device__ __forceinline__ unsigned pack_bf16(float a, float b) {
  union { __bf16 h[2]; unsigned u; } p;
  p.h[0] = (__bf16)a; p.h[1] = (__bf16)b;
  return p.u;
}

__device__ __forceinline__ float sigf(float x)     { return 1.0f / (1.0f + __expf(-x)); }
__device__ __forceinline__ float tanhfast(float x) { return 2.0f / (1.0f + __expf(-2.0f * x)) - 1.0f; }

// Grid-wide barrier across the persistent layer kernel's 64 blocks.
__device__ __forceinline__ void grid_barrier(unsigned* cnt, unsigned* gen) {
  __threadfence();
  __syncthreads();
  if (threadIdx.x == 0) {
    unsigned g = *(volatile unsigned*)gen;
    if (atomicAdd(cnt, 1u) == (unsigned)(NBLOCKS - 1)) {
      *(volatile unsigned*)cnt = 0u;
      __threadfence();
      atomicAdd(gen, 1u);
    } else {
      while (*(volatile unsigned*)gen == g) { __builtin_amdgcn_s_sleep(2); }
    }
  }
  __syncthreads();
  __threadfence();
}

// ---------------- small helper kernels ----------------
__global__ void init_barrier_kernel(unsigned* bar) {
  if (threadIdx.x < 2) bar[threadIdx.x] = 0u;
}

__global__ void convert_weights_kernel(const float* __restrict__ Wf,
                                       const float* __restrict__ Wo,
                                       __bf16* __restrict__ w0,
                                       __bf16* __restrict__ w12) {
  const int n0 = NWIDTH * K0;
  const int n1 = 2 * NWIDTH * K1;
  for (int i = blockIdx.x * blockDim.x + threadIdx.x; i < n0 + n1;
       i += gridDim.x * blockDim.x) {
    if (i < n0) w0[i] = (__bf16)Wf[i];
    else        w12[i - n0] = (__bf16)Wo[i - n0];
  }
}

// ---------------- persistent per-layer kernel ----------------
// Block b owns M rows [b*32, b*32+32). Wave w: mt = w&1 (16-row half-tile),
// hq = w>>1 (16-wide h slice). Wave computes gate[m-tile, g*128+hq*16 .. +16)
// for all 6 gates -> 6 v8f WMMA accumulators; the elementwise GRU-style
// update is then fully wave-local (no cross-wave traffic).
template <int LAYER, int KTOT>
__global__ __launch_bounds__(NTHREADS, 1)
void witran_layer_kernel(const float* __restrict__ inp,   // [B,R,O,F] (layer 0 only)
                         const __bf16* __restrict__ Wb,   // [768, KTOT] bf16
                         const float* __restrict__ Bias,  // [NL, 768]
                         const __bf16* __restrict__ aIn,  // [RB,L,256] bf16 (layers 1,2)
                         __bf16* __restrict__ aOutBf,     // [RB,L,256] bf16 (layers 0,1)
                         float* __restrict__ outF32,      // d_out base
                         float* __restrict__ hrow,        // [RB,H] f32 state
                         float* __restrict__ hcol,        // [2][RB,H] f32 state (double buffered)
                         unsigned* __restrict__ bar) {
  __shared__ __align__(16) __bf16 Abuf[32 * AS];        //  2.5 KB
  __shared__ __align__(16) __bf16 Bbuf[NWIDTH * BS];    // 60.0 KB

  const int tid    = threadIdx.x;
  const int lane   = tid & 31;
  const int w      = tid >> 5;
  const int laneLo = lane & 15;
  const int hi8    = (lane >> 4) << 3;    // A-frag K offset (bf16 elems) for lanes 16..31
  const int hi16   = (lane >> 4) << 4;    // unused; B frag uses hi8, same K order as A
  const int mt     = w & 1;
  const int hq     = w >> 1;
  const int m0     = blockIdx.x * 32;
  constexpr int nkc = KTOT >> 5;

  // --- per-layer state reset (h0 = zeros). Each block zeroes its own rows. ---
  for (int i = tid; i < 32 * HID; i += NTHREADS) {
    int mr = i >> 7, hh = i & 127;
    hrow[(m0 + mr) * HID + hh] = 0.0f;
    hcol[RB * HID + (m0 + mr) * HID + hh] = 0.0f;   // buf[1] is read at s=0
  }
  grid_barrier(bar, bar + 1);

  const float* bvec = Bias + LAYER * NWIDTH;
  const int h = (hq << 4) + laneLo;
  const float bb0 = bvec[0 * HID + h], bb1 = bvec[1 * HID + h];
  const float bb2 = bvec[2 * HID + h], bb3 = bvec[3 * HID + h];
  const float bb4 = bvec[4 * HID + h], bb5 = bvec[5 * HID + h];

  const uint4* Wq = (const uint4*)Wb;                 // 8 bf16 per uint4
  const unsigned* aInU = (const unsigned*)aIn;

#pragma unroll 1
  for (int s = 0; s < LSTEPS; ++s) {
    const float* hcolIn  = hcol + (((s + 1) & 1) ? (RB * HID) : 0);
    float*       hcolOut = hcol + (((s) & 1)     ? (RB * HID) : 0);

    v8f acc[6];
#pragma unroll
    for (int g = 0; g < 6; ++g)
#pragma unroll
      for (int j = 0; j < 8; ++j) acc[g][j] = 0.0f;

    // ---------------- K loop: z = [h_row | h_col(rolled) | a_s] ----------------
#pragma unroll 2
    for (int kc = 0; kc < nkc; ++kc) {
      __syncthreads();

      // Stage A tile: 32 rows x 32 k (one packed uint per thread).
      {
        int mr = tid >> 4, kp = tid & 15;
        int k = (kc << 5) + (kp << 1);
        int m = m0 + mr;
        unsigned pk;
        if (k < HID) {
          pk = pack_bf16(hrow[m * HID + k], hrow[m * HID + k + 1]);
        } else if (k < 2 * HID) {
          int mc = (m + RB - BATCH) & (RB - 1);        // rolled column hidden
          pk = pack_bf16(hcolIn[mc * HID + (k - HID)],
                         hcolIn[mc * HID + (k - HID + 1)]);
        } else if (LAYER == 0) {
          int g = m >> 6, b = m & 63, c = s - g, f = k - 2 * HID;
          float v0 = 0.0f, v1 = 0.0f;
          if ((unsigned)c < (unsigned)OCOLS) {
            const float* p = inp + (((b * RROWS + g) * OCOLS + c) * FIN + f);
            v0 = p[0]; v1 = p[1];
          }
          pk = pack_bf16(v0, v1);
        } else {
          pk = aInU[((m * LSTEPS + s) * 2 * HID + (k - 2 * HID)) >> 1];
        }
        *(unsigned*)&Abuf[mr * AS + (kp << 1)] = pk;
      }

      // Stage B tile: 768 n-rows x 32 k as uint4 (b128 global load -> b128 LDS store).
      for (int i = tid; i < NWIDTH * 4; i += NTHREADS) {
        int n = i >> 2, qc = i & 3;
        uint4 v = Wq[n * (KTOT >> 3) + (kc << 2) + qc];
        *(uint4*)&Bbuf[n * BS + (qc << 3)] = v;
        if (kc + 1 < nkc)  // prefetch next K-chunk of weights into cache
          __builtin_prefetch(&Wq[n * (KTOT >> 3) + ((kc + 1) << 2) + qc], 0, 1);
      }

      __syncthreads();

      // A fragment (16x32 bf16): lane -> M = lane%16; K per ISA 7.12.2.
      // Per-lane data = 2 contiguous 16B runs -> 2x ds_load_b128.
      FragU fa;
      {
        const __bf16* ap = &Abuf[(mt * 16 + laneLo) * AS + hi8];
        fa.q[0] = *(const uint4*)(ap);
        fa.q[1] = *(const uint4*)(ap + 16);
      }

      // 6 gate B fragments + WMMA. B (32x16): lane -> N = lane%16; K same as A (8 hi, +16 2nd run).
#pragma unroll
      for (int g = 0; g < 6; ++g) {
        FragU fb;
        const __bf16* bp = &Bbuf[(g * HID + (hq << 4) + laneLo) * BS + hi8];
        fb.q[0] = *(const uint4*)(bp);
        fb.q[1] = *(const uint4*)(bp + 16);
        acc[g] = __builtin_amdgcn_wmma_f32_16x16x32_bf16(
            false, fa.v, false, fb.v, (short)0, acc[g], false, false);
      }
    }

    // ---------------- fused elementwise update (wave-local) ----------------
#pragma unroll
    for (int j = 0; j < 8; ++j) {
      int m = m0 + mt * 16 + j + hi8;          // C layout: M = j + 8*(lane/16)
      int g = m >> 6, b = m & 63;
      float bm = ((g <= s) && (s < RROWS)) ? 1.0f : 0.0f;

      float ur  = sigf(acc[0][j] + bm * bb0);
      float orw = sigf(acc[1][j] + bm * bb1);
      float uc  = sigf(acc[2][j] + bm * bb2);
      float oc  = sigf(acc[3][j] + bm * bb3);
      float ir  = tanhfast(acc[4][j] + bm * bb4);
      float ic  = tanhfast(acc[5][j] + bm * bb5);

      float hr = hrow[m * HID + h];
      int   mc = (m + RB - BATCH) & (RB - 1);
      float hc = hcolIn[mc * HID + h];

      float hrn = tanhfast((1.0f - ur) * hr + ur * ir) * orw;
      float hcn = tanhfast((1.0f - uc) * hc + uc * ic) * oc;

      hrow[m * HID + h]    = hrn;
      hcolOut[m * HID + h] = hcn;

      int obase = (m * LSTEPS + s) * 2 * HID;
      if (LAYER < 2) {              // layers 0,1: bf16 activations for next layer
        aOutBf[obase + h]       = (__bf16)hrn;
        aOutBf[obase + HID + h] = (__bf16)hcn;
      } else {                      // layer 2: final output_all_slice (f32)
        outF32[obase + h]       = hrn;
        outF32[obase + HID + h] = hcn;
      }
      // hidden_row_all[b, layer, g, h]: row g finishes at slice O-1+g
      if (s >= OCOLS - 1 && g == s - (OCOLS - 1))
        outF32[ROW_SAVE_OFF + ((b * NL + LAYER) * RROWS + g) * HID + h] = hrn;
      // hidden_col_all[b, layer, s-(R-1), h]: last row group, pre-roll h_col
      if (g == RROWS - 1 && s >= RROWS - 1)
        outF32[COL_SAVE_OFF + ((b * NL + LAYER) * OCOLS + (s - (RROWS - 1))) * HID + h] = hcn;
    }

    grid_barrier(bar, bar + 1);
  }
}

// ---------------- host entry ----------------
extern "C" void kernel_launch(void* const* d_in, const int* in_sizes, int n_in,
                              void* d_out, int out_size, void* d_ws, size_t ws_size,
                              hipStream_t stream) {
  const float* inp     = (const float*)d_in[0];   // [B,R,O,F]
  const float* W_first = (const float*)d_in[1];   // [768, 288]
  const float* W_other = (const float*)d_in[2];   // [2, 768, 512]
  const float* Bias    = (const float*)d_in[3];   // [3, 768]
  float* out = (float*)d_out;

  char* ws = (char*)d_ws;
  size_t off = 0;
  unsigned* bar = (unsigned*)(ws + off);  off += 256;
  float* hrow   = (float*)(ws + off);     off += (size_t)RB * HID * 4;        // 1 MB
  float* hcol   = (float*)(ws + off);     off += (size_t)2 * RB * HID * 4;    // 2 MB
  __bf16* Wb0   = (__bf16*)(ws + off);    off += (size_t)NWIDTH * K0 * 2;     // 432 KB
  __bf16* Wb12  = (__bf16*)(ws + off);    off += (size_t)2 * NWIDTH * K1 * 2; // 1.5 MB
  __bf16* Abig  = (__bf16*)(ws + off);    off += (size_t)RB * LSTEPS * 2 * HID * 2; // 79 MB
  __bf16* Wb1 = Wb12;
  __bf16* Wb2 = Wb12 + (size_t)NWIDTH * K1;

  init_barrier_kernel<<<1, 64, 0, stream>>>(bar);
  convert_weights_kernel<<<512, 256, 0, stream>>>(W_first, W_other, Wb0, Wb12);

  // Layer 0: input from inp (wavefront shift computed on the fly), K=288.
  witran_layer_kernel<0, K0><<<NBLOCKS, NTHREADS, 0, stream>>>(
      inp, Wb0, Bias, nullptr, Abig, out, hrow, hcol, bar);
  // Layer 1: reads/writes Abig in place (per-slice, block-local rows), K=512.
  witran_layer_kernel<1, K1><<<NBLOCKS, NTHREADS, 0, stream>>>(
      inp, Wb1, Bias, Abig, Abig, out, hrow, hcol, bar);
  // Layer 2: reads Abig, writes final f32 slices + saves to d_out, K=512.
  witran_layer_kernel<2, K1><<<NBLOCKS, NTHREADS, 0, stream>>>(
      inp, Wb2, Bias, Abig, nullptr, out, hrow, hcol, bar);
}
